// NTCrossEntropyLoss_3702261809487
// MI455X (gfx1250) — matrix operand order repaired and checked by reading, not compiled, not run on hardware
//
#include <hip/hip_runtime.h>
#include <hip/hip_bf16.h>
#include <math.h>

typedef _Float16 v16h __attribute__((ext_vector_type(16)));
typedef _Float16 v8h  __attribute__((ext_vector_type(8)));
typedef float    v8f  __attribute__((ext_vector_type(8)));

#define MTOT 8192
#define NHALF 4096
#define DIM 512
#define LDSPAD 40   // halfs per LDS row (80B: 16B-aligned, bank-spread)

// ---------------------------------------------------------------------------
// Kernel 1: row-normalize concat(rep1,rep2) -> f16 X in workspace.
// One wave per row (32 lanes x 16 floats = 512).
// ---------------------------------------------------------------------------
__global__ __launch_bounds__(256)
void nt_normalize_kernel(const float* __restrict__ rep1,
                         const float* __restrict__ rep2,
                         _Float16* __restrict__ Xh) {
  const int row  = blockIdx.x * 8 + (threadIdx.x >> 5);
  const int lane = threadIdx.x & 31;
  const float* src = (row < NHALF) ? (rep1 + (size_t)row * DIM)
                                   : (rep2 + (size_t)(row - NHALF) * DIM);
  float vals[16];
  float ss = 0.0f;
#pragma unroll
  for (int i = 0; i < 16; ++i) {
    vals[i] = src[lane + i * 32];
    ss += vals[i] * vals[i];
  }
#pragma unroll
  for (int m = 16; m >= 1; m >>= 1) ss += __shfl_xor(ss, m, 32);
  const float inv = 1.0f / fmaxf(sqrtf(ss), 1e-8f);
  _Float16* dst = Xh + (size_t)row * DIM;
#pragma unroll
  for (int i = 0; i < 16; ++i) dst[lane + i * 32] = (_Float16)(vals[i] * inv);
}

// ---------------------------------------------------------------------------
// Kernel 2: zero the group accumulator G[8192].
// ---------------------------------------------------------------------------
__global__ void nt_zero_kernel(float* __restrict__ G) {
  G[blockIdx.x * 256 + threadIdx.x] = 0.0f;
}

// ---------------------------------------------------------------------------
// Kernel 3: 128x128 tile of S = X X^T via v_wmma_f32_16x16x32_f16.
// 8 waves (4x2 grid), each wave owns a 32x64 region = 2x4 WMMA tiles.
// Epilogue: per element, d=(c-jj)&8191; skip d==0; d==N -> Pos[jj]=s;
// else exp(2s) -> LDS histogram at exact group row r = (d_idx*M+jj)/(M-2),
// then flush nonzero slots with global f32 atomics.
// ---------------------------------------------------------------------------
__global__ __launch_bounds__(256)
void nt_simgemm_kernel(const _Float16* __restrict__ Xh,
                       float* __restrict__ G,
                       float* __restrict__ Pos) {
  __shared__ _Float16 As[128][LDSPAD];
  __shared__ _Float16 Bs[128][LDSPAD];
  __shared__ float    Gl[MTOT];

  const int R0 = blockIdx.y * 128;           // S row block (jj)
  const int C0 = blockIdx.x * 128;           // S col block (c)
  const int t    = threadIdx.x;
  const int lane = t & 31;
  const int wid  = t >> 5;
  const int wrow = wid >> 1;                 // 0..3  (32-row stripes)
  const int wcol = wid & 1;                  // 0..1  (64-col stripes)

  for (int i = t; i < MTOT; i += 256) Gl[i] = 0.0f;

  v8f acc[2][4];
#pragma unroll
  for (int i = 0; i < 2; ++i)
#pragma unroll
    for (int j = 0; j < 4; ++j)
#pragma unroll
      for (int e = 0; e < 8; ++e) acc[i][j][e] = 0.0f;

  // staging: 2 threads per row, 16 halfs (32B) each
  const int srow = t >> 1;
  const int sseg = t & 1;
  const size_t arow = (size_t)(R0 + srow) * DIM;
  const size_t brow = (size_t)(C0 + srow) * DIM;
  const int m = lane & 15;                   // row/col within 16-tile
  const int h = lane >> 4;                   // half-wave select

  for (int ks = 0; ks < DIM / 32; ++ks) {
    __syncthreads();
    const int kb = ks * 32;
    const uint4* ga = (const uint4*)(Xh + arow + kb) + sseg * 2;
    const uint4* gb = (const uint4*)(Xh + brow + kb) + sseg * 2;
    uint4 a0 = ga[0], a1 = ga[1];
    uint4 b0 = gb[0], b1 = gb[1];
    *((uint4*)&As[srow][sseg * 16])     = a0;
    *((uint4*)&As[srow][sseg * 16] + 1) = a1;
    *((uint4*)&Bs[srow][sseg * 16])     = b0;
    *((uint4*)&Bs[srow][sseg * 16] + 1) = b1;
    if (ks + 1 < DIM / 32) {                 // gfx1250 global_prefetch_b8
      __builtin_prefetch(Xh + arow + kb + 32, 0, 1);
      __builtin_prefetch(Xh + brow + kb + 32, 0, 1);
    }
    __syncthreads();

    // A fragments (ISA 7.12.2 16-bit A 16x32 layout)
    v16h afrag[2];
#pragma unroll
    for (int i = 0; i < 2; ++i) {
      const _Float16* ap = &As[wrow * 32 + i * 16 + m][0];
      v8h lo = *(const v8h*)(ap + h * 8);        // K = h*8 .. h*8+7
      v8h hi = *(const v8h*)(ap + 16 + h * 8);   // K = 16+h*8 ..
      afrag[i] = __builtin_shufflevector(lo, hi, 0,1,2,3,4,5,6,7,
                                                 8,9,10,11,12,13,14,15);
    }
    // B fragments: column N = lane%16, K = h*8..h*8+7 and 16+h*8..16+h*8+7
    v16h bfrag[4];
#pragma unroll
    for (int j = 0; j < 4; ++j) {
      const _Float16* bp = &Bs[wcol * 64 + j * 16 + m][h * 8];
      v8h lo = *(const v8h*)(bp);
      v8h hi = *(const v8h*)(bp + 16);
      bfrag[j] = __builtin_shufflevector(lo, hi, 0,1,2,3,4,5,6,7,
                                                 8,9,10,11,12,13,14,15);
    }
#pragma unroll
    for (int i = 0; i < 2; ++i)
#pragma unroll
      for (int j = 0; j < 4; ++j)
        acc[i][j] = __builtin_amdgcn_wmma_f32_16x16x32_f16(
            false, afrag[i], false, bfrag[j], (short)0, acc[i][j],
            false, false);
  }

  // Epilogue: C layout -> (jj, c); route exp(2s) to exact group row.
#pragma unroll
  for (int i = 0; i < 2; ++i) {
#pragma unroll
    for (int j = 0; j < 4; ++j) {
#pragma unroll
      for (int v = 0; v < 8; ++v) {
        const int jj = R0 + wrow * 32 + i * 16 + v + 8 * (lane >> 4);
        const int c  = C0 + wcol * 64 + j * 16 + (lane & 15);
        const float s = acc[i][j][v];
        const int d = (c - jj) & (MTOT - 1);
        if (d == 0) continue;                        // diagonal: excluded
        if (d == NHALF) {
          Pos[jj] = s;                               // positive pair (unique)
        } else {
          const unsigned didx = (unsigned)(d - 1 - (d > NHALF));
          const unsigned k = didx * (unsigned)MTOT + (unsigned)jj;
          const unsigned r = k / (unsigned)(MTOT - 2);
          atomicAdd(&Gl[r], __expf(2.0f * s));       // ds_add_f32
        }
      }
    }
  }
  __syncthreads();
  for (int i = t; i < MTOT; i += 256) {
    const float s = Gl[i];
    if (s != 0.0f) atomicAdd(&G[i], s);              // global_atomic_add_f32
  }
}

// ---------------------------------------------------------------------------
// Kernel 4: loss = (1/M) * sum_r [ log(G[r] + exp(2 Pos[r])) - 2 Pos[r] ]
// ---------------------------------------------------------------------------
__global__ __launch_bounds__(256)
void nt_reduce_kernel(const float* __restrict__ G,
                      const float* __restrict__ Pos,
                      float* __restrict__ out) {
  __shared__ float red[256];
  float acc = 0.0f;
  for (int r = threadIdx.x; r < MTOT; r += 256) {
    const float p2 = 2.0f * Pos[r];
    acc += logf(G[r] + __expf(p2)) - p2;
  }
  red[threadIdx.x] = acc;
  __syncthreads();
  for (int s = 128; s > 0; s >>= 1) {
    if (threadIdx.x < s) red[threadIdx.x] += red[threadIdx.x + s];
    __syncthreads();
  }
  if (threadIdx.x == 0) out[0] = red[0] / (float)MTOT;
}

// ---------------------------------------------------------------------------
extern "C" void kernel_launch(void* const* d_in, const int* in_sizes, int n_in,
                              void* d_out, int out_size, void* d_ws, size_t ws_size,
                              hipStream_t stream) {
  const float* rep1 = (const float*)d_in[0];
  const float* rep2 = (const float*)d_in[1];
  float* out = (float*)d_out;

  // workspace layout
  _Float16* Xh = (_Float16*)d_ws;                           // 8 MB
  float* G   = (float*)((char*)d_ws + (size_t)MTOT * DIM * 2);        // 32 KB
  float* Pos = (float*)((char*)d_ws + (size_t)MTOT * DIM * 2 + MTOT * 4);

  nt_normalize_kernel<<<MTOT / 8, 256, 0, stream>>>(rep1, rep2, Xh);
  nt_zero_kernel<<<MTOT / 256, 256, 0, stream>>>(G);
  nt_simgemm_kernel<<<dim3(MTOT / 128, MTOT / 128), 256, 0, stream>>>(Xh, G, Pos);
  nt_reduce_kernel<<<1, 256, 0, stream>>>(G, Pos, out);
}
